// MHSA_1743756722645
// MI455X (gfx1250) — matrix operand order repaired and checked by reading, not compiled, not run on hardware
//
#include <hip/hip_runtime.h>
#include <math.h>

#define HH  12
#define DIN 768
#define DD  64
#define NN  4
#define TT  1024

typedef __attribute__((ext_vector_type(16))) _Float16 v16h;
typedef __attribute__((ext_vector_type(8)))  float    v8f;
typedef __attribute__((ext_vector_type(4)))  _Float16 v4h;
typedef __attribute__((ext_vector_type(4)))  float    v4f;

// ---------------- WMMA fragment helpers (CDNA5 wave32 layouts) ----------------
// 16-bit A-matrix 16x32 (MxK), k order chosen by this code (not the ISA table's): lanes 0-15 hold M=0..15, K=0..15; lanes 16-31 K=16..31 (matches B).
__device__ __forceinline__ int a_kidx(int i, int half) {
    return half * 16 + 2 * i;
}

// A fragment from row-major f16 matrix; src points at (row0, k0), leading dim ld.
__device__ __forceinline__ v16h load_a_f16(const _Float16* src, int ld) {
    int lane = threadIdx.x & 31;
    int m = lane & 15, half = lane >> 4;
    const _Float16* p = src + (size_t)m * ld;
    v16h a;
#pragma unroll
    for (int i = 0; i < 8; ++i) {
        int k = a_kidx(i, half);
        a[2 * i]     = p[k];
        a[2 * i + 1] = p[k + 1];
    }
    return a;
}

// A fragment gathered from LDS f32 (probabilities), converted to f16.
__device__ __forceinline__ v16h load_a_lds(const float* src, int ld) {
    int lane = threadIdx.x & 31;
    int m = lane & 15, half = lane >> 4;
    const float* p = src + (size_t)m * ld;
    v16h a;
#pragma unroll
    for (int i = 0; i < 8; ++i) {
        int k = a_kidx(i, half);
        a[2 * i]     = (_Float16)p[k];
        a[2 * i + 1] = (_Float16)p[k + 1];
    }
    return a;
}

// B fragment (32x16, KxN) from row-major f16 [K, ld]; src at (k0, col0).
// lanes 0-15: N=lane, K=0..15 (2 per VGPR); lanes 16-31: N=lane-16, K=16..31.
__device__ __forceinline__ v16h load_b_f16(const _Float16* src, int ld) {
    int lane = threadIdx.x & 31;
    int nco = lane & 15, half = lane >> 4;
    v16h b;
#pragma unroll
    for (int j = 0; j < 8; ++j) {
        int k = half * 16 + 2 * j;
        b[2 * j]     = src[(size_t)k * ld + nco];
        b[2 * j + 1] = src[(size_t)(k + 1) * ld + nco];
    }
    return b;
}

// B fragment where memory holds B^T row-major: B(k,n) = src[n*ld + k].
__device__ __forceinline__ v16h load_bt_f16(const _Float16* src, int ld) {
    int lane = threadIdx.x & 31;
    int nco = lane & 15, half = lane >> 4;
    const _Float16* p = src + (size_t)nco * ld;
    v16h b;
#pragma unroll
    for (int j = 0; j < 8; ++j) {
        int k = half * 16 + 2 * j;
        b[2 * j]     = p[k];
        b[2 * j + 1] = p[k + 1];
    }
    return b;
}

__device__ __forceinline__ v8f wmma16(v16h a, v16h b, v8f c) {
    return __builtin_amdgcn_wmma_f32_16x16x32_f16(false, a, false, b, (short)0, c,
                                                  false, false);
}

// ---------------- CDNA5 async copy: global -> LDS (ASYNCcnt tracked) ----------
// Generic pointers to __shared__ carry the LDS byte offset in their low 32 bits
// (ISA 10.2 aperture rules), which is exactly what VDST of the async op wants.
__device__ __forceinline__ void async_copy_512B(void* lds_dst, const _Float16* gsrc,
                                                int n512) {
    int lane = threadIdx.x & 31;
    unsigned lbase = (unsigned)(uintptr_t)lds_dst + (unsigned)lane * 16u;
    const _Float16* g = gsrc + lane * 8;
    for (int i = 0; i < n512; ++i) {
        unsigned l = lbase + (unsigned)i * 512u;
        const _Float16* gp = g + i * 256;
        asm volatile("global_load_async_to_lds_b128 %0, %1, off"
                     :: "v"(l), "v"(gp) : "memory");
    }
}
__device__ __forceinline__ void wait_async(void) {
    asm volatile("s_wait_asynccnt 0x0" ::: "memory");
}
__device__ __forceinline__ void wait_async_le4(void) {
    asm volatile("s_wait_asynccnt 0x4" ::: "memory");
}
__device__ __forceinline__ void wait_async_le8(void) {
    asm volatile("s_wait_asynccnt 0x8" ::: "memory");
}

// ---------------- Kernel 0: f32 -> f16 convert (x4 vectorized) ----------------
__global__ void cvt_f16_kernel(const v4f* __restrict__ src,
                               v4h* __restrict__ dst, int count4) {
    int i = blockIdx.x * blockDim.x + threadIdx.x;
    if (i < count4) {
        v4f v = src[i];
        v4h o;
        o[0] = (_Float16)v.x; o[1] = (_Float16)v.y;
        o[2] = (_Float16)v.z; o[3] = (_Float16)v.w;
        dst[i] = o;
    }
}

// ---------------- Kernel 1: QKV projections ----------------
// grid = N*H*(T/16)*3, block = 32 (one wave). K-loop over DIN, A reused x4.
__global__ void qkv_proj_kernel(const _Float16* __restrict__ xh,
                                const _Float16* __restrict__ wqh,
                                const _Float16* __restrict__ wkh,
                                const _Float16* __restrict__ wvh,
                                _Float16* __restrict__ qh,
                                _Float16* __restrict__ kh,
                                _Float16* __restrict__ vh) {
    int b    = blockIdx.x;
    int proj = b % 3;
    int tile = (b / 3) % (TT / 16);
    int h    = (b / (3 * (TT / 16))) % HH;
    int n    = b / (3 * (TT / 16) * HH);

    const _Float16* w   = (proj == 0) ? wqh : (proj == 1) ? wkh : wvh;
    _Float16*       dst = (proj == 0) ? qh  : (proj == 1) ? kh  : vh;

    int r0 = tile * 16;
    const _Float16* xbase = xh + ((size_t)n * TT + r0) * DIN;
    const _Float16* wbase = w + (size_t)h * DIN * DD;

    v8f acc[4] = {};
    for (int k0 = 0; k0 < DIN; k0 += 32) {
        v16h a = load_a_f16(xbase + k0, DIN);
#pragma unroll
        for (int ct = 0; ct < 4; ++ct) {
            v16h bb = load_b_f16(wbase + (size_t)k0 * DD + ct * 16, DD);
            acc[ct] = wmma16(a, bb, acc[ct]);
        }
    }

    int lane = threadIdx.x & 31;
    int nco = lane & 15, half = lane >> 4;
    _Float16* dbase = dst + ((size_t)(n * HH + h) * TT + r0) * DD;
#pragma unroll
    for (int ct = 0; ct < 4; ++ct)
#pragma unroll
        for (int r = 0; r < 8; ++r)
            dbase[(size_t)(r + 8 * half) * DD + ct * 16 + nco] = (_Float16)acc[ct][r];
}

// ---------------- Kernel 2: fused causal attention ----------------
// grid = N*H*(T/16), block = 32. 16xT score strip in LDS; K/V tiles streamed in
// with double-buffered async-to-LDS copies; attn written to HBM exactly once
// with contiguous nontemporal b128 bursts.
__global__ void attn_kernel(const _Float16* __restrict__ qh,
                            const _Float16* __restrict__ kh,
                            const _Float16* __restrict__ vh,
                            float* __restrict__ attn,
                            _Float16* __restrict__ oh) {
    __shared__ v4f      sc4[16 * TT / 4];            // 64 KB score/prob strip
    __shared__ _Float16 kvbuf[2][32 * DD];           // 2 x 4 KB KV staging
    float* sc = (float*)sc4;

    int b    = blockIdx.x;
    int tile = b % (TT / 16);
    int h    = (b / (TT / 16)) % HH;
    int n    = b / ((TT / 16) * HH);
    int r0   = tile * 16;
    size_t nh = (size_t)(n * HH + h);

    const _Float16* qbase = qh + (nh * TT + r0) * DD;
    const _Float16* kbase = kh + nh * TT * DD;
    const _Float16* vbase = vh + nh * TT * DD;

    int lane = threadIdx.x & 31;
    int nco = lane & 15, half = lane >> 4;
    int ncols = (tile + 1) * 16;            // causal extent for this tile
    int ncp   = (ncols + 31) & ~31;         // padded to WMMA K granularity

    // ---- init: pad region of the strip = -inf so softmax can run full-T ----
    {
        v4f ninf = {-INFINITY, -INFINITY, -INFINITY, -INFINITY};
        for (int r = 0; r < 16; ++r) {
            v4f* srow4 = sc4 + r * (TT / 4);
            for (int c4 = ncols / 4 + lane; c4 < TT / 4; c4 += 32) srow4[c4] = ninf;
        }
    }

    // Q A-fragments are invariant across key tiles: load once.
    v16h aq0 = load_a_f16(qbase + 0, DD);
    v16h aq1 = load_a_f16(qbase + 32, DD);

    // ---- phase 1: S = Q K^T, masked, into LDS (K tiles double-buffered) ----
    async_copy_512B(kvbuf[0], kbase, 4);             // 16 keys x 64 f16 = 2 KB
    for (int s = 0; s <= tile; ++s) {
        bool more = (s < tile);
        if (more) async_copy_512B(kvbuf[(s + 1) & 1], kbase + (size_t)(s + 1) * 16 * DD, 4);
        if (more) wait_async_le4(); else wait_async();

        const _Float16* kt = kvbuf[s & 1];
        v8f c = {};
        c = wmma16(aq0, load_bt_f16(kt + 0, DD), c);
        c = wmma16(aq1, load_bt_f16(kt + 32, DD), c);

        int col = s * 16 + nco;
#pragma unroll
        for (int r = 0; r < 8; ++r) {
            int m   = r + 8 * half;
            int row = r0 + m;
            float v = c[r];
            // reference: upper triangle -> 0 -> -inf; any exact 0 -> -inf
            if (col > row || v == 0.0f) v = -INFINITY;
            sc[m * TT + col] = v;
        }
    }
    // Start streaming the first V chunk while softmax runs.
    async_copy_512B(kvbuf[0], vbase, 8);             // 32 keys x 64 f16 = 4 KB
    __syncthreads();

    // ---- phase 2: row-serial exact softmax; coalesced NT b128 attn writes ----
    for (int r = 0; r < 16; ++r) {
        v4f* srow4 = sc4 + r * (TT / 4);
        float mx = -INFINITY;
        for (int c4 = lane; c4 < TT / 4; c4 += 32) {
            v4f v = srow4[c4];
            mx = fmaxf(mx, fmaxf(fmaxf(v.x, v.y), fmaxf(v.z, v.w)));
        }
#pragma unroll
        for (int off = 16; off > 0; off >>= 1) mx = fmaxf(mx, __shfl_xor(mx, off, 32));

        float sum = 0.0f;
        for (int c4 = lane; c4 < TT / 4; c4 += 32) {
            v4f v = srow4[c4];
            v.x = __expf(v.x - mx); v.y = __expf(v.y - mx);
            v.z = __expf(v.z - mx); v.w = __expf(v.w - mx);
            srow4[c4] = v;
            sum += (v.x + v.y) + (v.z + v.w);
        }
#pragma unroll
        for (int off = 16; off > 0; off >>= 1) sum += __shfl_xor(sum, off, 32);
        float inv = 1.0f / sum;

        v4f* arow4 = (v4f*)(attn + (nh * TT + (size_t)(r0 + r)) * TT);
        for (int c4 = lane; c4 < TT / 4; c4 += 32) {
            v4f v = srow4[c4];
            v *= inv;
            srow4[c4] = v;                               // probs for AV
            __builtin_nontemporal_store(v, arow4 + c4);  // streamed, never reread
        }
    }
    __syncthreads();

    // ---- phase 3: O = P V (V chunks double-buffered via async-to-LDS) ----
    v8f acc[4] = {};
    for (int kc = 0; kc < ncp; kc += 32) {
        int cur = (kc >> 5) & 1;
        bool more = (kc + 32) < ncp;
        if (more) async_copy_512B(kvbuf[cur ^ 1], vbase + (size_t)(kc + 32) * DD, 8);
        if (more) wait_async_le8(); else wait_async();

        const _Float16* vt = kvbuf[cur];
        v16h a = load_a_lds(sc + kc, TT);
#pragma unroll
        for (int ct = 0; ct < 4; ++ct) {
            v16h bb = load_b_f16(vt + ct * 16, DD);
            acc[ct] = wmma16(a, bb, acc[ct]);
        }
    }
    _Float16* obase = oh + (nh * TT + r0) * DD;
#pragma unroll
    for (int ct = 0; ct < 4; ++ct)
#pragma unroll
        for (int r = 0; r < 8; ++r)
            obase[(size_t)(r + 8 * half) * DD + ct * 16 + nco] = (_Float16)acc[ct][r];
}

// ---------------- Kernel 3: output projection out = concat(o) @ Wo + bo ----
__global__ void out_proj_kernel(const _Float16* __restrict__ oh,
                                const _Float16* __restrict__ woh,
                                const float* __restrict__ bo,
                                float* __restrict__ out) {
    int b    = blockIdx.x;
    int tile = b % (TT / 16);
    int n    = b / (TT / 16);
    int r0   = tile * 16;

    int lane = threadIdx.x & 31;
    int nco = lane & 15, half = lane >> 4;
    int ma = lane & 15;

    v8f acc[4] = {};
    for (int k0 = 0; k0 < HH * DD; k0 += 32) {
        v16h a;
#pragma unroll
        for (int i = 0; i < 8; ++i) {
            int k = k0 + a_kidx(i, half);   // k even -> k,k+1 share one head
            int hh = k >> 6, dd = k & 63;
            const _Float16* p = oh + ((size_t)(n * HH + hh) * TT + r0 + ma) * DD;
            a[2 * i]     = p[dd];
            a[2 * i + 1] = p[dd + 1];
        }
#pragma unroll
        for (int ct = 0; ct < 4; ++ct) {
            v16h bb = load_b_f16(woh + (size_t)k0 * DD + ct * 16, DD);
            acc[ct] = wmma16(a, bb, acc[ct]);
        }
    }

    float* obase = out + ((size_t)n * TT + r0) * DD;
#pragma unroll
    for (int ct = 0; ct < 4; ++ct)
#pragma unroll
        for (int r = 0; r < 8; ++r) {
            int col = ct * 16 + nco;
            obase[(size_t)(r + 8 * half) * DD + col] = acc[ct][r] + bo[col];
        }
}

// ---------------- Launch ----------------
extern "C" void kernel_launch(void* const* d_in, const int* in_sizes, int n_in,
                              void* d_out, int out_size, void* d_ws, size_t ws_size,
                              hipStream_t stream) {
    const float* x  = (const float*)d_in[0];
    const float* Wq = (const float*)d_in[1];
    const float* Wk = (const float*)d_in[2];
    const float* Wv = (const float*)d_in[3];
    const float* Wo = (const float*)d_in[4];
    const float* bo = (const float*)d_in[5];

    float* out  = (float*)d_out;
    float* attn = out + (size_t)NN * TT * DD;   // outputs concatenated: (out, attn)

    // Workspace carve-up (f16 elements)
    _Float16* ws = (_Float16*)d_ws;
    size_t off = 0;
    _Float16* xh  = ws + off; off += (size_t)NN * TT * DIN;
    _Float16* wqh = ws + off; off += (size_t)HH * DIN * DD;
    _Float16* wkh = ws + off; off += (size_t)HH * DIN * DD;
    _Float16* wvh = ws + off; off += (size_t)HH * DIN * DD;
    _Float16* woh = ws + off; off += (size_t)HH * DD * DD;
    _Float16* qh  = ws + off; off += (size_t)NN * HH * TT * DD;
    _Float16* kh  = ws + off; off += (size_t)NN * HH * TT * DD;
    _Float16* vh  = ws + off; off += (size_t)NN * HH * TT * DD;
    _Float16* oh  = ws + off; off += (size_t)NN * HH * TT * DD;

    auto cvt = [&](const float* s, _Float16* d, size_t cnt) {
        size_t c4 = cnt / 4;
        cvt_f16_kernel<<<(unsigned)((c4 + 255) / 256), 256, 0, stream>>>(
            (const v4f*)s, (v4h*)d, (int)c4);
    };
    cvt(x,  xh,  (size_t)NN * TT * DIN);
    cvt(Wq, wqh, (size_t)HH * DIN * DD);
    cvt(Wk, wkh, (size_t)HH * DIN * DD);
    cvt(Wv, wvh, (size_t)HH * DIN * DD);
    cvt(Wo, woh, (size_t)HH * DD * DD);

    qkv_proj_kernel<<<NN * HH * (TT / 16) * 3, 32, 0, stream>>>(xh, wqh, wkh, wvh,
                                                                qh, kh, vh);
    attn_kernel<<<NN * HH * (TT / 16), 32, 0, stream>>>(qh, kh, vh, attn, oh);
    out_proj_kernel<<<NN * (TT / 16), 32, 0, stream>>>(oh, woh, bo, out);
}
